// Latent_1331439862076
// MI455X (gfx1250) — matrix operand order repaired and checked by reading, not compiled, not run on hardware
//
#include <hip/hip_runtime.h>
#include <hip/hip_bf16.h>

typedef __attribute__((ext_vector_type(16))) __bf16 v16bf;
typedef __attribute__((ext_vector_type(8)))  __bf16 v8bf;
typedef __attribute__((ext_vector_type(8)))  float  v8f;

#define N_ 16384
#define M_ 4096
#define D_ 256
#define TJ 32
#define WISE_BLOCKS (M_ / TJ)  // 128

// ---- order-preserving float<->uint encoding so atomicMin(uint) == float min ----
__device__ __forceinline__ unsigned f2ord(float f) {
    unsigned u = __float_as_uint(f);
    return (u & 0x80000000u) ? ~u : (u | 0x80000000u);
}
__device__ __forceinline__ float ord2f(unsigned u) {
    unsigned v = (u & 0x80000000u) ? (u & 0x7fffffffu) : ~u;
    return __uint_as_float(v);
}

// colmin_u[j] = f2ord(+inf) = 0xFF800000
__global__ void init_colmin(unsigned* __restrict__ colmin_u) {
    colmin_u[blockIdx.x * 256 + threadIdx.x] = 0xFF800000u;
}

// Split fp32 rows into bf16 hi/lo pair + row sum of squares. One block (256 thr) per row (D_=256).
__global__ void __launch_bounds__(256)
split_bf16_rows(const float* __restrict__ x, __bf16* __restrict__ xh,
                __bf16* __restrict__ xl, float* __restrict__ x2) {
    __shared__ float red[256];
    const int row = blockIdx.x, t = threadIdx.x;
    const size_t idx = (size_t)row * D_ + t;
    float f = x[idx];
    __bf16 h = (__bf16)f;
    float hf = (float)h;
    __bf16 l = (__bf16)(f - hf);
    xh[idx] = h;
    xl[idx] = l;
    red[t] = f * f;
    __syncthreads();
    for (int s = 128; s > 0; s >>= 1) {
        if (t < s) red[t] += red[t + s];
        __syncthreads();
    }
    if (t == 0) x2[row] = red[0];
}

// WMMA GEMM (bf16x3 emulated fp32) + running column-min of s[i,j] = z2[i] - 2*(z.e).
// Block = 128 threads = 4 waves; wave w handles j-tile (blockIdx.x*4+w)*16, i-range blockIdx.y*128.
__global__ void __launch_bounds__(128)
gemm_colmin(const __bf16* __restrict__ zh, const __bf16* __restrict__ zl,
            const __bf16* __restrict__ eh, const __bf16* __restrict__ el,
            const float* __restrict__ z2, unsigned* __restrict__ colmin_u) {
    const int lane = threadIdx.x & 31;
    const int wave = threadIdx.x >> 5;
    const int j0 = (blockIdx.x * 4 + wave) * 16;
    const int iBase = blockIdx.y * 128;
    const int l15 = lane & 15;
    const int half = lane >> 4;

    // B fragment base: lane n -> column j0+n, contiguous K; upper half-wave offset +16 in K
    const __bf16* bRowH = eh + (size_t)(j0 + l15) * D_ + half * 16;
    const __bf16* bRowL = el + (size_t)(j0 + l15) * D_ + half * 16;

    float m = __builtin_inff();

    for (int it = 0; it < 8; ++it) {
        const int i0 = iBase + it * 16;
        // A fragment base: lane l15 -> row i0+l15; halves at K {0..15} / {16..31}
        const __bf16* aRowH = zh + (size_t)(i0 + l15) * D_ + half * 16;
        const __bf16* aRowL = zl + (size_t)(i0 + l15) * D_ + half * 16;
        v8f acc = {};
#pragma unroll
        for (int ks = 0; ks < 8; ++ks) {
            const int k0 = ks * 32;
            union Frag { v16bf v; v8bf h[2]; };
            Frag ah, al, bh, bl;
            ah.h[0] = *(const v8bf*)(aRowH + k0);
            ah.h[1] = *(const v8bf*)(aRowH + k0 + 8);
            al.h[0] = *(const v8bf*)(aRowL + k0);
            al.h[1] = *(const v8bf*)(aRowL + k0 + 8);
            bh.v = *(const v16bf*)(bRowH + k0);
            bl.v = *(const v16bf*)(bRowL + k0);
            // fp32 emulation: acc += zl*eh + zh*el + zh*eh
            acc = __builtin_amdgcn_wmma_f32_16x16x32_bf16(false, al.v, false, bh.v,
                                                          (short)0, acc, false, false);
            acc = __builtin_amdgcn_wmma_f32_16x16x32_bf16(false, ah.v, false, bl.v,
                                                          (short)0, acc, false, false);
            acc = __builtin_amdgcn_wmma_f32_16x16x32_bf16(false, ah.v, false, bh.v,
                                                          (short)0, acc, false, false);
        }
        // C layout: VGPR r, lane -> element (i0 + r + 8*half, j0 + l15)
        const int ib = i0 + half * 8;
#pragma unroll
        for (int r = 0; r < 8; ++r) {
            float s = z2[ib + r] - 2.0f * acc[r];
            m = fminf(m, s);
        }
    }
    // combine the two half-wave row groups for the same column j
    m = fminf(m, __shfl_xor(m, 16, 32));
    if (lane < 16) atomicMin(colmin_u + (j0 + l15), f2ord(m));
}

// wise_min: each block owns TJ columns of e (in registers) x all 256 d (one per lane),
// streams z rows from L2; writes one deterministic partial sum per block.
__global__ void __launch_bounds__(256)
wise_min_partial(const float* __restrict__ z, const float* __restrict__ e,
                 float* __restrict__ partial) {
    __shared__ float red[256];
    const int d = threadIdx.x;
    const int jb = blockIdx.x * TJ;
    float er[TJ], acc[TJ];
#pragma unroll
    for (int jj = 0; jj < TJ; ++jj) {
        er[jj] = e[(size_t)(jb + jj) * D_ + d];
        acc[jj] = __builtin_inff();
    }
    const float* zp = z + d;
    for (int i = 0; i < N_; i += 4) {
        if (i + 16 < N_) __builtin_prefetch(zp + (size_t)(i + 16) * D_, 0, 0);
        float zv0 = zp[(size_t)(i + 0) * D_];
        float zv1 = zp[(size_t)(i + 1) * D_];
        float zv2 = zp[(size_t)(i + 2) * D_];
        float zv3 = zp[(size_t)(i + 3) * D_];
#pragma unroll
        for (int jj = 0; jj < TJ; ++jj) {
            float d0 = zv0 - er[jj];
            float d1 = zv1 - er[jj];
            float d2 = zv2 - er[jj];
            float d3 = zv3 - er[jj];
            float m01 = fminf(d0 * d0, d1 * d1);
            float m23 = fminf(d2 * d2, d3 * d3);
            acc[jj] = fminf(acc[jj], fminf(m01, m23));
        }
    }
    float s = 0.f;
#pragma unroll
    for (int jj = 0; jj < TJ; ++jj) s += acc[jj];
    red[d] = s;
    __syncthreads();
    for (int t = 128; t > 0; t >>= 1) {
        if (d < t) red[d] += red[d + t];
        __syncthreads();
    }
    if (d == 0) partial[blockIdx.x] = red[0];
}

// Single-block deterministic finalize: both scalar outputs.
__global__ void __launch_bounds__(256)
finalize(const float* __restrict__ e2, const unsigned* __restrict__ colmin_u,
         const float* __restrict__ partial, float* __restrict__ out) {
    __shared__ float r1[256], r2[256];
    const int t = threadIdx.x;
    float s1 = 0.f, s2 = 0.f;
    for (int j = t; j < M_; j += 256) s1 += e2[j] + ord2f(colmin_u[j]);
    for (int b = t; b < WISE_BLOCKS; b += 256) s2 += partial[b];
    r1[t] = s1;
    r2[t] = s2;
    __syncthreads();
    for (int s = 128; s > 0; s >>= 1) {
        if (t < s) {
            r1[t] += r1[t + s];
            r2[t] += r2[t + s];
        }
        __syncthreads();
    }
    if (t == 0) {
        out[0] = r1[0] / (float)M_;
        out[1] = r2[0] / ((float)M_ * (float)D_);
    }
}

extern "C" void kernel_launch(void* const* d_in, const int* in_sizes, int n_in,
                              void* d_out, int out_size, void* d_ws, size_t ws_size,
                              hipStream_t stream) {
    const float* z = (const float*)d_in[0];  // [N, D] fp32
    const float* e = (const float*)d_in[1];  // [M, D] fp32
    float* out = (float*)d_out;              // [2] fp32: min_loss, wise_min_loss
    char* ws = (char*)d_ws;

    // workspace layout (bytes), ~21.1 MB total
    __bf16* zh = (__bf16*)(ws + 0);                 // N*D*2 = 8 MB
    __bf16* zl = (__bf16*)(ws + 8388608);           // 8 MB
    __bf16* eh = (__bf16*)(ws + 16777216);          // M*D*2 = 2 MB
    __bf16* el = (__bf16*)(ws + 18874368);          // 2 MB
    float* z2 = (float*)(ws + 20971520);            // N*4
    float* e2 = (float*)(ws + 21037056);            // M*4
    unsigned* colmin = (unsigned*)(ws + 21053440);  // M*4
    float* partial = (float*)(ws + 21069824);       // WISE_BLOCKS*4

    init_colmin<<<M_ / 256, 256, 0, stream>>>(colmin);
    split_bf16_rows<<<N_, 256, 0, stream>>>(z, zh, zl, z2);
    split_bf16_rows<<<M_, 256, 0, stream>>>(e, eh, el, e2);
    gemm_colmin<<<dim3(M_ / 64, N_ / 128), 128, 0, stream>>>(zh, zl, eh, el, z2, colmin);
    wise_min_partial<<<WISE_BLOCKS, 256, 0, stream>>>(z, e, partial);
    finalize<<<1, 256, 0, stream>>>(e2, colmin, partial, out);
}
